// GNN_VAE_25245817766163
// MI455X (gfx1250) — compile-verified, matrix operand order checked by reading, not run on hardware
//
#include <hip/hip_runtime.h>
#include <hip/hip_bf16.h>

// ---------------------------------------------------------------------------
// GNN-VAE forward for MI455X (gfx1250, wave32).
// Dense GEMMs (gat, gcn, mu, logvar, res, decoder-MLP) run on
// v_wmma_f32_16x16x32_bf16 with f32 accumulation; irregular graph phases
// (edge softmax, message aggregation, degree norm) use float atomics.
// LDS tiles use a 56-element (112B) row stride: 16B-aligned rows (so staging
// and fragment accesses vectorize to ds_*_b128) and conflict-free bank
// mapping for the 16-lane fragment reads (28n mod 64 distinct for n<16).
// ---------------------------------------------------------------------------

typedef __attribute__((ext_vector_type(16))) __bf16 v16bf;
typedef __attribute__((ext_vector_type(8)))  float  v8f;

#define D_IN 256
#define HID  128
#define LAT  64
#define DEC  256
#define LDL  56   // LDS row stride in bf16 elements (112 bytes)

// ---------------- helpers ----------------

__device__ __forceinline__ unsigned fenc(float f) {
  unsigned u = __float_as_uint(f);
  return (u & 0x80000000u) ? ~u : (u | 0x80000000u);
}
__device__ __forceinline__ float fdec(unsigned u) {
  u = (u & 0x80000000u) ? (u & 0x7fffffffu) : ~u;
  return __uint_as_float(u);
}
__device__ __forceinline__ float lrelu(float e) { return e >= 0.f ? e : 0.2f * e; }

// 16x32 bf16 fragment from row-major LDS tile (row stride LDL).
// ISA layout: lanes 0-15 -> row=lane, elems 0..7 = K(h*8..), 8..15 = K(16+h*8..)
__device__ __forceinline__ v16bf frag_rowmajor(const __bf16* s, int lane) {
  int m = lane & 15, h = (lane >> 4) & 1;
  const __bf16* r = s + (size_t)m * LDL;
  v16bf a;
#pragma unroll
  for (int i = 0; i < 8; ++i) a[i] = r[h * 8 + i];
#pragma unroll
  for (int i = 0; i < 8; ++i) a[8 + i] = r[16 + h * 8 + i];
  return a;
}

// ---------------- generic WMMA GEMM: C[M,N] = A[M,K] @ W[K,N] (+bias) -------
// block = 256 threads (8 waves); tile = 32(M) x 64(N); K stepped by 32.
// W staged transposed (n-major) in LDS so B-fragment reads are contiguous.

__global__ void wmma_gemm(const float* __restrict__ A, const float* __restrict__ W,
                          const float* __restrict__ bias, float* __restrict__ C,
                          int M, int K, int N, int relu) {
  __shared__ __align__(16) __bf16 As[32][LDL];
  __shared__ __align__(16) __bf16 Bs[64][LDL];
  const int t = threadIdx.x;
  const int lane = t & 31;
  const int wave = t >> 5;
  const int mt = wave >> 2;   // 0..1
  const int nt = wave & 3;    // 0..3
  const int m0 = blockIdx.y * 32;
  const int n0 = blockIdx.x * 64;
  v8f acc = {};

  for (int kk = 0; kk < K; kk += 32) {
    // stage A tile (32x32 f32 -> bf16), 4 contiguous elems/thread -> ds b64
    {
      int m = t >> 3, kc = (t & 7) * 4;
      float4 v = make_float4(0.f, 0.f, 0.f, 0.f);
      int gm = m0 + m;
      if (gm < M) v = *(const float4*)(A + (size_t)gm * K + kk + kc);
      As[m][kc + 0] = (__bf16)v.x; As[m][kc + 1] = (__bf16)v.y;
      As[m][kc + 2] = (__bf16)v.z; As[m][kc + 3] = (__bf16)v.w;
    }
    // stage W tile transposed: Bs[n][k] = W[kk+k][n0+n].
    // Each thread: fixed n, 8 contiguous k -> one 16B ds store; global reads
    // coalesced over n for each k.
    {
      int n = t & 63, k0 = (t >> 6) * 8;
      float tmp[8];
#pragma unroll
      for (int j = 0; j < 8; ++j)
        tmp[j] = W[(size_t)(kk + k0 + j) * N + n0 + n];
#pragma unroll
      for (int j = 0; j < 8; ++j)
        Bs[n][k0 + j] = (__bf16)tmp[j];
    }
    __syncthreads();
    v16bf fa = frag_rowmajor(&As[mt * 16][0], lane);
    v16bf fb = frag_rowmajor(&Bs[nt * 16][0], lane);
    acc = __builtin_amdgcn_wmma_f32_16x16x32_bf16(false, fa, false, fb,
                                                  (short)0, acc, false, false);
    __syncthreads();
  }

  int h = lane >> 4;
  int col = n0 + nt * 16 + (lane & 15);
  float bv = bias ? bias[col] : 0.f;
#pragma unroll
  for (int r = 0; r < 8; ++r) {
    int row = m0 + mt * 16 + h * 8 + r;
    if (row < M) {
      float v = acc[r] + bv;
      if (relu) v = fmaxf(v, 0.f);
      C[(size_t)row * N + col] = v;
    }
  }
}

// ---------------- decoder: recon[e] = W2 . relu([resz[row]||resz[col]] @ W1 + b1) + b2

__global__ void decoder_kernel(const float* __restrict__ resz,
                               const int* __restrict__ rowI, const int* __restrict__ colI,
                               const __bf16* __restrict__ W1T, // [256][128] n-major bf16
                               const float* __restrict__ b1, const float* __restrict__ W2,
                               const float* __restrict__ b2, float* __restrict__ recon,
                               int E) {
  __shared__ __align__(16) __bf16 As[32][LDL];
  __shared__ __align__(16) __bf16 Bs[256][LDL];
  __shared__ float red[32];
  const int t = threadIdx.x;
  const int lane = t & 31;
  const int wave = t >> 5;
  const int mt = wave >> 2;       // 0..1 -> rows mt*16..
  const int ng = wave & 3;        // col group: cols ng*64 .. +63
  const int e0 = blockIdx.x * 32;

  if (t < 32) red[t] = 0.f;

  v8f acc[4] = {v8f{}, v8f{}, v8f{}, v8f{}};

  for (int kk = 0; kk < 2 * LAT; kk += 32) {
    // gather A: row m = [resz[rowI[e]] (k<64) || resz[colI[e]] (k>=64)]
    {
      int m = t >> 3, kc = (t & 7) * 4;
      int e = e0 + m;
      int node = 0;
      if (e < E) node = (kk < LAT) ? rowI[e] : colI[e];
      int off = (kk & (LAT - 1)) + kc;
      float4 v = *(const float4*)(resz + (size_t)node * LAT + off);
      As[m][kc + 0] = (__bf16)v.x; As[m][kc + 1] = (__bf16)v.y;
      As[m][kc + 2] = (__bf16)v.z; As[m][kc + 3] = (__bf16)v.w;
    }
    // stage W1T chunk: Bs[n][k] = W1T[n][kk+k]; 32 contiguous bf16 per
    // thread (n = t): 64B global load + 4x 16B LDS stores.
    {
      const __bf16* src = W1T + (size_t)t * (2 * LAT) + kk;
#pragma unroll
      for (int i = 0; i < 32; ++i) Bs[t][i] = src[i];
    }
    __syncthreads();
    v16bf fa = frag_rowmajor(&As[mt * 16][0], lane);
#pragma unroll
    for (int j = 0; j < 4; ++j) {
      v16bf fb = frag_rowmajor(&Bs[ng * 64 + j * 16][0], lane);
      acc[j] = __builtin_amdgcn_wmma_f32_16x16x32_bf16(false, fa, false, fb,
                                                       (short)0, acc[j], false, false);
    }
    __syncthreads();
  }

  // fused epilogue: bias + relu + dot with W2, per-row LDS reduction
  int h = lane >> 4;
  float prow[8];
#pragma unroll
  for (int r = 0; r < 8; ++r) prow[r] = 0.f;
#pragma unroll
  for (int j = 0; j < 4; ++j) {
    int col = ng * 64 + j * 16 + (lane & 15);
    float bb = b1[col], w2 = W2[col];
#pragma unroll
    for (int r = 0; r < 8; ++r) {
      float v = fmaxf(acc[j][r] + bb, 0.f);
      prow[r] += v * w2;
    }
  }
#pragma unroll
  for (int r = 0; r < 8; ++r) {
    int rloc = mt * 16 + h * 8 + r;
    atomicAdd(&red[rloc], prow[r]);
  }
  __syncthreads();
  if (t < 32) {
    int e = e0 + t;
    if (e < E) recon[e] = red[t] + b2[0];
  }
}

// ---------------- graph / elementwise kernels ----------------

__global__ void fill_f32(float* p, float v, long n) {
  long i = (long)blockIdx.x * blockDim.x + threadIdx.x;
  if (i < n) p[i] = v;
}
__global__ void fill_u32(unsigned* p, unsigned v, long n) {
  long i = (long)blockIdx.x * blockDim.x + threadIdx.x;
  if (i < n) p[i] = v;
}

__global__ void conv_w1t(const float* __restrict__ W1, __bf16* __restrict__ W1T) {
  int i = blockIdx.x * blockDim.x + threadIdx.x;   // over 128*256
  if (i < (2 * LAT) * DEC) {
    int k = i >> 8, n = i & 255;                   // W1 is [128][256]
    W1T[(size_t)n * (2 * LAT) + k] = (__bf16)W1[i];
  }
}

// a_src[n,h] = xp[n,h,:].att_src[h,:]; one wave per (n,h)
__global__ void att_scores(const float* __restrict__ xp, const float* __restrict__ as,
                           const float* __restrict__ ad, float* __restrict__ a_src,
                           float* __restrict__ a_dst, int N) {
  int g = (blockIdx.x * blockDim.x + threadIdx.x) >> 5;
  int lane = threadIdx.x & 31;
  if (g >= N * 2) return;
  int n = g >> 1, h = g & 1;
  const float* xr = xp + (size_t)n * D_IN + h * HID;
  float s = 0.f, d = 0.f;
#pragma unroll
  for (int i = 0; i < 4; ++i) {
    float xv = xr[lane + 32 * i];
    s += xv * as[h * HID + lane + 32 * i];
    d += xv * ad[h * HID + lane + 32 * i];
  }
#pragma unroll
  for (int o = 16; o > 0; o >>= 1) {
    s += __shfl_down(s, o, 32);
    d += __shfl_down(d, o, 32);
  }
  if (lane == 0) { a_src[n * 2 + h] = s; a_dst[n * 2 + h] = d; }
}

__global__ void att_max(const int* __restrict__ srcI, const int* __restrict__ dstI,
                        const float* __restrict__ a_src, const float* __restrict__ a_dst,
                        unsigned* __restrict__ nmax, int E, int N) {
  int i = blockIdx.x * blockDim.x + threadIdx.x;
  if (i >= E + N) return;
  int s = i < E ? srcI[i] : i - E;
  int d = i < E ? dstI[i] : i - E;
#pragma unroll
  for (int h = 0; h < 2; ++h) {
    float e = lrelu(a_src[s * 2 + h] + a_dst[d * 2 + h]);
    atomicMax(&nmax[d * 2 + h], fenc(e));
  }
}

__global__ void decode_max(unsigned* __restrict__ u, int n) {
  int i = blockIdx.x * blockDim.x + threadIdx.x;
  if (i < n) u[i] = __float_as_uint(fdec(u[i]));
}

__global__ void att_expsum(const int* __restrict__ srcI, const int* __restrict__ dstI,
                           const float* __restrict__ a_src, const float* __restrict__ a_dst,
                           const float* __restrict__ nmax, float* __restrict__ ex,
                           float* __restrict__ denom, int E, int N) {
  int i = blockIdx.x * blockDim.x + threadIdx.x;
  if (i >= E + N) return;
  int s = i < E ? srcI[i] : i - E;
  int d = i < E ? dstI[i] : i - E;
#pragma unroll
  for (int h = 0; h < 2; ++h) {
    float e = lrelu(a_src[s * 2 + h] + a_dst[d * 2 + h]);
    float x = __expf(e - nmax[d * 2 + h]);
    ex[(size_t)i * 2 + h] = x;
    atomicAdd(&denom[d * 2 + h], x);
  }
}

// one wave per edge; 4 features per lane; heads averaged
__global__ void att_agg(const int* __restrict__ srcI, const int* __restrict__ dstI,
                        const float* __restrict__ xp, const float* __restrict__ ex,
                        const float* __restrict__ denom, float* __restrict__ h1,
                        int E, int N) {
  long gid = (long)blockIdx.x * blockDim.x + threadIdx.x;
  int lane = (int)(gid & 31);
  long i = gid >> 5;
  if (i >= (long)E + N) return;
  int s = i < E ? srcI[i] : (int)(i - E);
  int d = i < E ? dstI[i] : (int)(i - E);
  float al0 = ex[i * 2 + 0] / denom[d * 2 + 0];
  float al1 = ex[i * 2 + 1] / denom[d * 2 + 1];
  float4 x0 = *(const float4*)(xp + (size_t)s * D_IN + lane * 4);
  float4 x1 = *(const float4*)(xp + (size_t)s * D_IN + HID + lane * 4);
  float* o = h1 + (size_t)d * HID + lane * 4;
  atomicAdd(o + 0, 0.5f * (x0.x * al0 + x1.x * al1));
  atomicAdd(o + 1, 0.5f * (x0.y * al0 + x1.y * al1));
  atomicAdd(o + 2, 0.5f * (x0.z * al0 + x1.z * al1));
  atomicAdd(o + 3, 0.5f * (x0.w * al0 + x1.w * al1));
}

__global__ void bias_relu(float* __restrict__ p, const float* __restrict__ b, int F, long n) {
  long i = (long)blockIdx.x * blockDim.x + threadIdx.x;
  if (i < n) p[i] = fmaxf(p[i] + b[i % F], 0.f);
}

__global__ void bn_stats(const float* __restrict__ x, float* __restrict__ mean,
                         float* __restrict__ rstd, int N, int F) {
  __shared__ float sh[256], sh2[256];
  int f = blockIdx.x;
  float s = 0.f, ss = 0.f;
  for (int i = threadIdx.x; i < N; i += 256) {
    float v = x[(size_t)i * F + f];
    s += v; ss += v * v;
  }
  sh[threadIdx.x] = s; sh2[threadIdx.x] = ss;
  __syncthreads();
  for (int o = 128; o > 0; o >>= 1) {
    if (threadIdx.x < o) { sh[threadIdx.x] += sh[threadIdx.x + o]; sh2[threadIdx.x] += sh2[threadIdx.x + o]; }
    __syncthreads();
  }
  if (threadIdx.x == 0) {
    float m = sh[0] / N;
    float var = sh2[0] / N - m * m;
    mean[f] = m;
    rstd[f] = rsqrtf(var + 1e-5f);
  }
}

__global__ void bn_apply(const float* __restrict__ x, const float* __restrict__ mean,
                         const float* __restrict__ rstd, const float* __restrict__ g,
                         const float* __restrict__ b, float* __restrict__ y, int F, long n) {
  long i = (long)blockIdx.x * blockDim.x + threadIdx.x;
  if (i < n) {
    int f = (int)(i % F);
    y[i] = (x[i] - mean[f]) * rstd[f] * g[f] + b[f];
  }
}

__global__ void gcn_deg(const int* __restrict__ dstI, float* __restrict__ deg, int E) {
  int i = blockIdx.x * blockDim.x + threadIdx.x;
  if (i < E) atomicAdd(&deg[dstI[i]], 1.0f);
}
__global__ void gcn_dinv(float* __restrict__ d, int N) {
  int i = blockIdx.x * blockDim.x + threadIdx.x;
  if (i < N) d[i] = rsqrtf(d[i]);   // deg >= 1 (self loop)
}

// 16 lanes per edge, 4 feats/lane
__global__ void gcn_agg(const int* __restrict__ srcI, const int* __restrict__ dstI,
                        const float* __restrict__ xp2, const float* __restrict__ dinv,
                        float* __restrict__ h2, int E, int N) {
  long gid = (long)blockIdx.x * blockDim.x + threadIdx.x;
  int lane = (int)(gid & 15);
  long i = gid >> 4;
  if (i >= (long)E + N) return;
  int s = i < E ? srcI[i] : (int)(i - E);
  int d = i < E ? dstI[i] : (int)(i - E);
  float nrm = dinv[s] * dinv[d];
  float4 x = *(const float4*)(xp2 + (size_t)s * LAT + lane * 4);
  float* o = h2 + (size_t)d * LAT + lane * 4;
  atomicAdd(o + 0, x.x * nrm);
  atomicAdd(o + 1, x.y * nrm);
  atomicAdd(o + 2, x.z * nrm);
  atomicAdd(o + 3, x.w * nrm);
}

__global__ void reparam(const float* __restrict__ mu, const float* __restrict__ lv,
                        const float* __restrict__ eps, float* __restrict__ z, long n) {
  long i = (long)blockIdx.x * blockDim.x + threadIdx.x;
  if (i < n) z[i] = mu[i] + eps[i] * __expf(0.5f * lv[i]);
}

// ---------------- launch ----------------

static inline int cdiv(long a, long b) { return (int)((a + b - 1) / b); }

extern "C" void kernel_launch(void* const* d_in, const int* in_sizes, int n_in,
                              void* d_out, int out_size, void* d_ws, size_t ws_size,
                              hipStream_t stream) {
  const float* x        = (const float*)d_in[0];
  const int*   edge     = (const int*)d_in[1];
  const float* gat_W    = (const float*)d_in[2];
  const float* gat_asrc = (const float*)d_in[3];
  const float* gat_adst = (const float*)d_in[4];
  const float* gat_b    = (const float*)d_in[5];
  const float* bn1_g    = (const float*)d_in[6];
  const float* bn1_b    = (const float*)d_in[7];
  const float* gcn_W    = (const float*)d_in[8];
  const float* gcn_b    = (const float*)d_in[9];
  const float* bn2_g    = (const float*)d_in[10];
  const float* bn2_b    = (const float*)d_in[11];
  const float* mu_W     = (const float*)d_in[12];
  const float* mu_b     = (const float*)d_in[13];
  const float* lv_W     = (const float*)d_in[14];
  const float* lv_b     = (const float*)d_in[15];
  const float* res_W    = (const float*)d_in[16];
  const float* res_b    = (const float*)d_in[17];
  const float* dec_W1   = (const float*)d_in[18];
  const float* dec_b1   = (const float*)d_in[19];
  const float* dec_W2   = (const float*)d_in[20];
  const float* dec_b2   = (const float*)d_in[21];
  const float* eps      = (const float*)d_in[22];

  const int N = in_sizes[0] / D_IN;
  const int E = in_sizes[1] / 2;
  const int Et = E + N;
  const int* srcI = edge;
  const int* dstI = edge + E;

  float* outF   = (float*)d_out;
  float* recon  = outF;               // [E]
  float* mu     = outF + E;           // [N*64]
  float* logvar = outF + E + (size_t)N * LAT;

  // workspace layout (floats)
  float* ws = (float*)d_ws;
  size_t o = 0;
  float*    xp    = ws + o; o += (size_t)N * D_IN;
  float*    a_src = ws + o; o += (size_t)N * 2;
  float*    a_dst = ws + o; o += (size_t)N * 2;
  unsigned* nmax  = (unsigned*)(ws + o); o += (size_t)N * 2;
  float*    denom = ws + o; o += (size_t)N * 2;
  float*    exbuf = ws + o; o += (size_t)Et * 2;
  float*    h1    = ws + o; o += (size_t)N * HID;
  float*    h1n   = ws + o; o += (size_t)N * HID;
  float*    xp2   = ws + o; o += (size_t)N * LAT;
  float*    deg   = ws + o; o += (size_t)N;
  float*    dinv  = ws + o; o += (size_t)N;
  float*    h2    = ws + o; o += (size_t)N * LAT;
  float*    h2n   = ws + o; o += (size_t)N * LAT;
  float*    zbuf  = ws + o; o += (size_t)N * LAT;
  float*    resz  = ws + o; o += (size_t)N * LAT;
  float*    mean1 = ws + o; o += HID;
  float*    rstd1 = ws + o; o += HID;
  float*    mean2 = ws + o; o += LAT;
  float*    rstd2 = ws + o; o += LAT;
  __bf16*   w1t   = (__bf16*)(ws + o); o += (size_t)(2 * LAT) * DEC / 2;

  const int T = 256;

  // ---- GAT ----
  wmma_gemm<<<dim3(D_IN / 64, cdiv(N, 32)), T, 0, stream>>>(x, gat_W, nullptr, xp, N, D_IN, D_IN, 0);
  att_scores<<<cdiv((long)N * 2 * 32, T), T, 0, stream>>>(xp, gat_asrc, gat_adst, a_src, a_dst, N);
  fill_u32<<<cdiv((long)N * 2, T), T, 0, stream>>>(nmax, 0u, (long)N * 2);
  fill_f32<<<cdiv((long)N * 2, T), T, 0, stream>>>(denom, 0.f, (long)N * 2);
  fill_f32<<<cdiv((long)N * HID, T), T, 0, stream>>>(h1, 0.f, (long)N * HID);
  att_max<<<cdiv(Et, T), T, 0, stream>>>(srcI, dstI, a_src, a_dst, nmax, E, N);
  decode_max<<<cdiv((long)N * 2, T), T, 0, stream>>>(nmax, N * 2);
  att_expsum<<<cdiv(Et, T), T, 0, stream>>>(srcI, dstI, a_src, a_dst, (const float*)nmax, exbuf, denom, E, N);
  att_agg<<<cdiv((long)Et * 32, T), T, 0, stream>>>(srcI, dstI, xp, exbuf, denom, h1, E, N);
  bias_relu<<<cdiv((long)N * HID, T), T, 0, stream>>>(h1, gat_b, HID, (long)N * HID);
  bn_stats<<<HID, T, 0, stream>>>(h1, mean1, rstd1, N, HID);
  bn_apply<<<cdiv((long)N * HID, T), T, 0, stream>>>(h1, mean1, rstd1, bn1_g, bn1_b, h1n, HID, (long)N * HID);

  // ---- GCN ----
  wmma_gemm<<<dim3(LAT / 64, cdiv(N, 32)), T, 0, stream>>>(h1n, gcn_W, nullptr, xp2, N, HID, LAT, 0);
  fill_f32<<<cdiv((long)N, T), T, 0, stream>>>(deg, 1.0f, (long)N);   // self-loop
  fill_f32<<<cdiv((long)N * LAT, T), T, 0, stream>>>(h2, 0.f, (long)N * LAT);
  gcn_deg<<<cdiv(E, T), T, 0, stream>>>(dstI, deg, E);
  gcn_dinv<<<cdiv(N, T), T, 0, stream>>>(deg, N);   // in-place: deg -> dinv
  gcn_agg<<<cdiv((long)Et * 16, T), T, 0, stream>>>(srcI, dstI, xp2, deg, h2, E, N);
  bias_relu<<<cdiv((long)N * LAT, T), T, 0, stream>>>(h2, gcn_b, LAT, (long)N * LAT);
  bn_stats<<<LAT, T, 0, stream>>>(h2, mean2, rstd2, N, LAT);
  bn_apply<<<cdiv((long)N * LAT, T), T, 0, stream>>>(h2, mean2, rstd2, bn2_g, bn2_b, h2n, LAT, (long)N * LAT);

  // ---- heads ----
  wmma_gemm<<<dim3(1, cdiv(N, 32)), T, 0, stream>>>(h2n, mu_W, mu_b, mu, N, LAT, LAT, 0);
  wmma_gemm<<<dim3(1, cdiv(N, 32)), T, 0, stream>>>(h2n, lv_W, lv_b, logvar, N, LAT, LAT, 0);
  reparam<<<cdiv((long)N * LAT, T), T, 0, stream>>>(mu, logvar, eps, zbuf, (long)N * LAT);
  wmma_gemm<<<dim3(1, cdiv(N, 32)), T, 0, stream>>>(zbuf, res_W, res_b, resz, N, LAT, LAT, 0);

  // ---- decoder (dominant GEMM, fused MLP) ----
  conv_w1t<<<cdiv((2 * LAT) * DEC, T), T, 0, stream>>>(dec_W1, w1t);
  decoder_kernel<<<cdiv(E, 32), T, 0, stream>>>(resz, srcI, dstI, w1t, dec_b1, dec_W2, dec_b2, recon, E);

  (void)n_in; (void)out_size; (void)ws_size;
}
